// DocREModel_56779467653404
// MI455X (gfx1250) — matrix operand order repaired and checked by reading, not compiled, not run on hardware
//
#include <hip/hip_runtime.h>

// ---------------------------------------------------------------------------
// DocRE pipeline for MI455X (gfx1250, wave32, WMMA).
// All dense GEMMs run on v_wmma_f32_16x16x32_bf16 (fp32 accum), 4 accumulators
// per wave (16x64 output tile), copy-free ping-pong double buffering (manual
// 2x k-unroll): loads of one buffer are issued, then 4 WMMAs consume the
// other, so each load has a full WMMA phase of latency cover.
// The 201MB "bl" bilinear tensor is never materialized: its 16x32 A-tiles are
// generated in-register from zh/zt outer products inside the final WMMA GEMM,
// with the VALU synthesis overlapping the in-flight B loads.
// B operands are pre-transposed to column-major bf16 so each lane's 8 VGPRs
// fill with two aligned global_load_b128 per operand tile.
// ---------------------------------------------------------------------------

#define DEV_INLINE __device__ __forceinline__

typedef unsigned short u16;
typedef unsigned int   u32;

typedef __attribute__((ext_vector_type(16))) __bf16 v16bf;
typedef __attribute__((ext_vector_type(8)))  float  v8f;

#define NN   4
#define CC   1024
#define DD   768
#define HH   12
#define EE   32
#define MMM  4
#define PPP  256
#define BSZ  64
#define NLL  97
#define NP   (NN*PPP)      // 1024
#define K3   (3*DD)        // 2304
#define KBIG (DD*BSZ)      // 49152

DEV_INLINE u32 f2bf(float f) {                 // round-to-nearest-even, bits
  u32 u = __float_as_uint(f);
  return (u + 0x7FFFu + ((u >> 16) & 1u)) >> 16;
}
DEV_INLINE float bf2f(u32 bits16) { return __uint_as_float(bits16 << 16); }

union ABPack { v16bf v; u32 u[8]; };

// A: 16x32 bf16 operand layout -- lane m = lane&15, kgrp = lane>>4,
// VGPR v holds K pair at 2v + 8*kgrp (+8 if v>=4).
DEV_INLINE void loadA(ABPack& a, const u16* __restrict__ Ap, int kb, int kg) {
#pragma unroll
  for (int v = 0; v < 8; ++v) {
    int ka = kb + 2 * v + 8 * kg + ((v >= 4) ? 8 : 0);
    a.u[v] = *(const u32*)(Ap + ka);
  }
}
// B: 32x16 bf16 operand layout -- lane n = lane&15, VGPR v holds K pair at
// 2v + 8*kgrp (+8 if v>=4), read from column-major Bt (K contiguous per column).
DEV_INLINE void loadB4(ABPack* b, const u16* __restrict__ const* Bp, int kb, int kg) {
#pragma unroll
  for (int i = 0; i < 4; ++i)
#pragma unroll
    for (int v = 0; v < 8; ++v)
      b[i].u[v] = *(const u32*)(Bp[i] + kb + 2 * v + 8 * kg + ((v >= 4) ? 8 : 0));
}

DEV_INLINE float block_reduce_sum(float v, float* sd) {
  int t = threadIdx.x;
  sd[t] = v; __syncthreads();
  for (int s = blockDim.x >> 1; s > 0; s >>= 1) {
    if (t < s) sd[t] += sd[t + s];
    __syncthreads();
  }
  float r = sd[0]; __syncthreads();
  return r;
}

// ------------------------- small prep kernels ------------------------------

__global__ void k_cvt_bf16(const float* __restrict__ src, u16* __restrict__ dst, long n) {
  long i = (long)blockIdx.x * blockDim.x + threadIdx.x;
  if (i < n) dst[i] = (u16)f2bf(src[i]);
}

// Bt[n*K + k] = B[k*N + n]   (column-major bf16 for WMMA B operand)
__global__ void k_tconv(const float* __restrict__ B, u16* __restrict__ Bt, int K, int N) {
  long i = (long)blockIdx.x * blockDim.x + threadIdx.x;
  if (i >= (long)K * N) return;
  int n = (int)(i / K), k = (int)(i % K);
  Bt[i] = (u16)f2bf(B[(long)k * N + n]);
}

// seqT[n][d][c] = seq[n][c][d]  (bf16)
__global__ void k_seqT(const float* __restrict__ seq, u16* __restrict__ seqT) {
  long i = (long)blockIdx.x * blockDim.x + threadIdx.x;
  if (i >= (long)NN * CC * DD) return;
  int n = (int)(i / (CC * DD));
  int r = (int)(i % (CC * DD));
  int d = r / CC, c = r % CC;
  seqT[i] = (u16)f2bf(seq[(long)n * CC * DD + (long)c * DD + d]);
}

// w1[d] = sum_j Wq[d,j]*aw[j];  w1[DD] = bq.aw1 + ab
__global__ void k_w1(const float* __restrict__ Wq, const float* __restrict__ bq,
                     const float* __restrict__ aw, const float* __restrict__ ab,
                     float* __restrict__ w1) {
  int d = blockIdx.x * blockDim.x + threadIdx.x;
  if (d < DD) {
    float s = 0.f;
    for (int j = 0; j < DD; ++j) s += Wq[(long)d * DD + j] * aw[j];
    w1[d] = s;
  } else if (d == DD) {
    float c = 0.f;
    for (int j = 0; j < DD; ++j) c += bq[j] * aw[j];
    w1[DD] = c + ab[0];
  }
}

// ent_emb[n,e,d] = logsumexp_m seq[n, pos[n,e,m], d]
__global__ void k_ent_emb(const float* __restrict__ seq, const int* __restrict__ pos,
                          float* __restrict__ ent_emb) {
  long i = (long)blockIdx.x * blockDim.x + threadIdx.x;
  if (i >= (long)NN * EE * DD) return;
  int n = (int)(i / (EE * DD));
  int r = (int)(i % (EE * DD));
  int e = r / DD, d = r % DD;
  float v[MMM], mx = -3.4e38f;
  for (int m = 0; m < MMM; ++m) {
    int p = pos[((long)n * EE + e) * MMM + m];
    v[m] = seq[((long)n * CC + p) * DD + d];
    mx = fmaxf(mx, v[m]);
  }
  float s = 0.f;
  for (int m = 0; m < MMM; ++m) s += __expf(v[m] - mx);
  ent_emb[i] = mx + __logf(s);
}

// ent_att[n,e,h,c] = mean_m attn[n,h,pos[n,e,m],c]
__global__ void k_ent_att(const float* __restrict__ attn, const int* __restrict__ pos,
                          float* __restrict__ ent_att) {
  long i = (long)blockIdx.x * blockDim.x + threadIdx.x;
  if (i >= (long)NN * EE * HH * CC) return;
  int c = (int)(i % CC);
  long t = i / CC;
  int h = (int)(t % HH); t /= HH;
  int e = (int)(t % EE);
  int n = (int)(t / EE);
  float s = 0.f;
  for (int m = 0; m < MMM; ++m) {
    int p = pos[((long)n * EE + e) * MMM + m];
    s += attn[(((long)n * HH + h) * CC + p) * CC + c];
  }
  ent_att[i] = s * (1.f / MMM);
}

// ht_bf[np, c] (bf16) = normalize_c( mean_h h_att*t_att ).  One block per (n,p).
__global__ void k_ht(const float* __restrict__ ent_att, const int* __restrict__ hts,
                     u16* __restrict__ ht_bf) {
  __shared__ float sd[256];
  int np = blockIdx.x;
  int n = np / PPP;
  int hi = hts[(long)np * 2 + 0];
  int ti = hts[(long)np * 2 + 1];
  const float* ha = ent_att + ((long)n * EE + hi) * HH * CC;
  const float* ta = ent_att + ((long)n * EE + ti) * HH * CC;
  float vals[4], part = 0.f;
  for (int i = 0; i < 4; ++i) {
    int c = threadIdx.x + 256 * i;
    float a = 0.f;
    for (int h = 0; h < HH; ++h)
      a += ha[(long)h * CC + c] * ta[(long)h * CC + c];
    vals[i] = a * (1.f / HH);
    part += vals[i];
  }
  float tot = block_reduce_sum(part, sd);
  float norm = 1.f / (tot + 1e-5f);
  for (int i = 0; i < 4; ++i)
    ht_bf[(long)np * CC + threadIdx.x + 256 * i] = (u16)f2bf(vals[i] * norm);
}

// row element-offsets into seq for the sk GEMM gather (both sides)
__global__ void k_arow(const int* __restrict__ pos, const int* __restrict__ hts,
                       int* __restrict__ arow_h, int* __restrict__ arow_t) {
  int i = blockIdx.x * blockDim.x + threadIdx.x;
  if (i >= 2 * NN * PPP * MMM) return;
  int side = i / (NN * PPP * MMM);
  int r = i % (NN * PPP * MMM);
  int n = r / (PPP * MMM);
  int pm = r % (PPP * MMM);
  int p = pm / MMM, m = pm % MMM;
  int idx = hts[((long)n * PPP + p) * 2 + side];
  int pv = pos[((long)n * EE + idx) * MMM + m];
  int off = (n * CC + pv) * DD;
  if (side == 0) arow_h[r] = off; else arow_t[r] = off;
}

// ------------------------- generic WMMA bf16 GEMM --------------------------
// C[M x Ncols] = act( A[M x K] @ B[K x Ncols] + bias ),  A row-major bf16
// (optional per-row gather), Bt column-major bf16 (Bt[col*ldk + k]).
// One wave per 16x64 output tile (4 accumulators), ping-pong k-loop.
// M%16==0, Ncols%64==0, K%64==0.
__global__ void k_wmma_gemm(const u16* __restrict__ A, const int* __restrict__ Arows,
                            int lda, long Astride,
                            const u16* __restrict__ Bt, int ldk, long Bstride,
                            const float* __restrict__ bias,
                            float* __restrict__ C, u16* __restrict__ Cbf, long Cstride,
                            int M, int Ncols, int K, int act, int tilesNg) {
  int wave = blockIdx.x * (blockDim.x >> 5) + (threadIdx.x >> 5);
  int lane = threadIdx.x & 31;
  int mt = wave / tilesNg, ng = wave % tilesNg;
  if (mt * 16 >= M) return;
  int n16 = lane & 15, kg = lane >> 4;

  long ab = (long)blockIdx.y * Astride;
  long bb_ = (long)blockIdx.y * Bstride;
  long cb = (long)blockIdx.y * Cstride;

  int arow = mt * 16 + n16;
  long rowStart = Arows ? (long)Arows[arow] : (long)arow * lda;
  const u16* Ap = A + ab + rowStart;
  const u16* Bp[4];
#pragma unroll
  for (int i = 0; i < 4; ++i)
    Bp[i] = Bt + bb_ + (long)(ng * 64 + i * 16 + n16) * ldk;

  v8f acc[4] = {};
  ABPack a0, a1, b0[4], b1[4];
  loadA(a0, Ap, 0, kg);
  loadB4(b0, Bp, 0, kg);

  int kb = 0;
  for (; kb + 64 < K; kb += 64) {
    // phase 0: issue buf1 loads @kb+32, compute with buf0
    loadA(a1, Ap, kb + 32, kg);
    loadB4(b1, Bp, kb + 32, kg);
    __builtin_prefetch(Ap + kb + 128, 0, 3);
    __builtin_prefetch(Bp[0] + kb + 128, 0, 3);
    __builtin_prefetch(Bp[1] + kb + 128, 0, 3);
    __builtin_prefetch(Bp[2] + kb + 128, 0, 3);
    __builtin_prefetch(Bp[3] + kb + 128, 0, 3);
#pragma unroll
    for (int i = 0; i < 4; ++i)
      acc[i] = __builtin_amdgcn_wmma_f32_16x16x32_bf16(false, a0.v, false, b0[i].v,
                                                       (short)0, acc[i], false, false);
    // phase 1: issue buf0 loads @kb+64, compute with buf1
    loadA(a0, Ap, kb + 64, kg);
    loadB4(b0, Bp, kb + 64, kg);
#pragma unroll
    for (int i = 0; i < 4; ++i)
      acc[i] = __builtin_amdgcn_wmma_f32_16x16x32_bf16(false, a1.v, false, b1[i].v,
                                                       (short)0, acc[i], false, false);
  }
  // epilogue: kb+64 == K
  loadA(a1, Ap, kb + 32, kg);
  loadB4(b1, Bp, kb + 32, kg);
#pragma unroll
  for (int i = 0; i < 4; ++i)
    acc[i] = __builtin_amdgcn_wmma_f32_16x16x32_bf16(false, a0.v, false, b0[i].v,
                                                     (short)0, acc[i], false, false);
#pragma unroll
  for (int i = 0; i < 4; ++i)
    acc[i] = __builtin_amdgcn_wmma_f32_16x16x32_bf16(false, a1.v, false, b1[i].v,
                                                     (short)0, acc[i], false, false);

#pragma unroll
  for (int i = 0; i < 4; ++i) {
    int col = ng * 64 + i * 16 + n16;
    float bv = bias ? bias[col] : 0.f;
#pragma unroll
    for (int v = 0; v < 8; ++v) {
      int r = mt * 16 + v + 8 * kg;                          // C/D layout: M = v + 8*kgrp
      float x = acc[i][v] + bv;
      if (act == 1) x = tanhf(x);
      else if (act == 2) x = tanhf(tanhf(x));
      if (C)   C[cb + (long)r * Ncols + col] = x;
      if (Cbf) Cbf[cb + (long)r * Ncols + col] = (u16)f2bf(x);
    }
  }
}

// --------------- mention-guide score/softmax/weighted-sum ------------------
// score_m = leakyrelu(rs.w1 + c1 + sk_m.aw2);  p = softmax_m;  ess = sum p_m sk_m
__global__ void k_score(const float* __restrict__ rs, const float* __restrict__ sk,
                        const float* __restrict__ w1, const float* __restrict__ aw,
                        const int* __restrict__ pos, const int* __restrict__ hts,
                        int side, float* __restrict__ ess) {
  __shared__ float sd[256];
  __shared__ float sc_s[MMM];
  __shared__ float pm_s[MMM];
  int np = blockIdx.x;
  const float* rsr = rs + (long)np * DD;

  float part = 0.f;
  for (int d = threadIdx.x; d < DD; d += 256) part += rsr[d] * w1[d];
  float rsa = block_reduce_sum(part, sd);

  for (int m = 0; m < MMM; ++m) {
    const float* skr = sk + ((long)np * MMM + m) * DD;
    float p2 = 0.f;
    for (int d = threadIdx.x; d < DD; d += 256) p2 += skr[d] * aw[DD + d];
    float sm = block_reduce_sum(p2, sd);
    if (threadIdx.x == 0) sc_s[m] = sm;
  }
  __syncthreads();

  if (threadIdx.x == 0) {
    int n = np / PPP;
    int idx = hts[(long)np * 2 + side];
    float s[MMM], mx = -3.4e38f;
    for (int m = 0; m < MMM; ++m) {
      float v = rsa + w1[DD] + sc_s[m];
      v = v < 0.f ? 0.01f * v : v;                 // leaky_relu 0.01
      int pv = pos[((long)n * EE + idx) * MMM + m];
      if (pv == 0) v = -1e9f;
      s[m] = v; mx = fmaxf(mx, v);
    }
    float tot = 0.f;
    for (int m = 0; m < MMM; ++m) { s[m] = __expf(s[m] - mx); tot += s[m]; }
    for (int m = 0; m < MMM; ++m) pm_s[m] = s[m] / tot;
  }
  __syncthreads();

  for (int d = threadIdx.x; d < DD; d += 256) {
    float o = 0.f;
    for (int m = 0; m < MMM; ++m) o += pm_s[m] * sk[((long)np * MMM + m) * DD + d];
    ess[(long)np * DD + d] = o;
  }
}

// cat = bf16([rs, ent_emb[n,idx], ess])  (NP x 2304)
__global__ void k_cat(const float* __restrict__ rs, const float* __restrict__ ent_emb,
                      const float* __restrict__ ess, const int* __restrict__ hts,
                      int side, u16* __restrict__ cat) {
  long i = (long)blockIdx.x * blockDim.x + threadIdx.x;
  if (i >= (long)NP * K3) return;
  int np = (int)(i / K3);
  int col = (int)(i % K3);
  int n = np / PPP;
  float v;
  if (col < DD) v = rs[(long)np * DD + col];
  else if (col < 2 * DD) {
    int idx = hts[(long)np * 2 + side];
    v = ent_emb[((long)n * EE + idx) * DD + (col - DD)];
  } else v = ess[(long)np * DD + (col - 2 * DD)];
  cat[i] = (u16)f2bf(v);
}

// ---------------- fused bilinear + classifier WMMA GEMM --------------------
// out[np,l] = sum_k bl[np,k]*Wb[k,l] + bb ;  bl[np, j*4096+x*64+y] = zh[np,j*64+x]*zt[np,j*64+y]
// A-tiles (16x32 bf16) synthesized in-register per k-step and reused by
// 4 WMMAs; ping-pong B buffers; synthesis VALU overlaps in-flight B loads.
__global__ void k_bilinear_out(const u16* __restrict__ zh, const u16* __restrict__ zt,
                               const u16* __restrict__ WbT, const float* __restrict__ bbv,
                               float* __restrict__ out) {
  const int nGroups = 2;                                     // cols [0,64) and [64,97)
  int wave = blockIdx.x * (blockDim.x >> 5) + (threadIdx.x >> 5);
  int lane = threadIdx.x & 31;
  int mt = wave / nGroups, ng = wave % nGroups;
  if (mt * 16 >= NP) return;
  int n16 = lane & 15, kg = lane >> 4;

  int row = mt * 16 + n16;
  const u16* zhr = zh + (long)row * DD;
  const u16* ztr = zt + (long)row * DD;
  const u16* Bp[4];
  int cols[4];
#pragma unroll
  for (int i = 0; i < 4; ++i) {
    int col = ng * 64 + i * 16 + n16;
    cols[i] = col;
    int colc = col < NLL ? col : NLL - 1;                    // clamp loads for tail tile
    Bp[i] = WbT + (long)colc * KBIG;
  }

  // synthesize one 16x32 bf16 A-tile for k-block kb
  auto synthA = [&](ABPack& a, int kb) {
    int j = kb >> 12;
    int x = (kb >> 6) & 63;
    int y0 = kb & 63;                                        // 0 or 32
    float s = bf2f(zhr[j * 64 + x]);
#pragma unroll
    for (int v = 0; v < 8; ++v) {
      int ka = 2 * v + 8 * kg + ((v >= 4) ? 8 : 0);          // K offset in [0,31]
      u32 zz = *(const u32*)(ztr + j * 64 + y0 + ka);        // two zt bf16
      float z0 = bf2f(zz & 0xFFFFu);
      float z1 = bf2f(zz >> 16);
      a.u[v] = f2bf(s * z0) | (f2bf(s * z1) << 16);
    }
  };

  v8f acc[4] = {};
  ABPack a, b0[4], b1[4];
  loadB4(b0, Bp, 0, kg);

  int kb = 0;
  for (; kb + 64 < KBIG; kb += 64) {
    // phase 0: issue buf1 loads @kb+32; synth A (overlaps loads); compute buf0
    loadB4(b1, Bp, kb + 32, kg);
    __builtin_prefetch(Bp[0] + kb + 128, 0, 3);
    __builtin_prefetch(Bp[1] + kb + 128, 0, 3);
    __builtin_prefetch(Bp[2] + kb + 128, 0, 3);
    __builtin_prefetch(Bp[3] + kb + 128, 0, 3);
    synthA(a, kb);
#pragma unroll
    for (int i = 0; i < 4; ++i)
      acc[i] = __builtin_amdgcn_wmma_f32_16x16x32_bf16(false, a.v, false, b0[i].v,
                                                       (short)0, acc[i], false, false);
    // phase 1: issue buf0 loads @kb+64; synth A; compute buf1
    loadB4(b0, Bp, kb + 64, kg);
    synthA(a, kb + 32);
#pragma unroll
    for (int i = 0; i < 4; ++i)
      acc[i] = __builtin_amdgcn_wmma_f32_16x16x32_bf16(false, a.v, false, b1[i].v,
                                                       (short)0, acc[i], false, false);
  }
  // epilogue: kb+64 == KBIG
  loadB4(b1, Bp, kb + 32, kg);
  synthA(a, kb);
#pragma unroll
  for (int i = 0; i < 4; ++i)
    acc[i] = __builtin_amdgcn_wmma_f32_16x16x32_bf16(false, a.v, false, b0[i].v,
                                                     (short)0, acc[i], false, false);
  synthA(a, kb + 32);
#pragma unroll
  for (int i = 0; i < 4; ++i)
    acc[i] = __builtin_amdgcn_wmma_f32_16x16x32_bf16(false, a.v, false, b1[i].v,
                                                     (short)0, acc[i], false, false);

#pragma unroll
  for (int i = 0; i < 4; ++i) {
    if (cols[i] < NLL) {
      float bv = bbv[cols[i]];
#pragma unroll
      for (int v = 0; v < 8; ++v) {
        int r = mt * 16 + v + 8 * kg;
        out[(long)r * NLL + cols[i]] = acc[i][v] + bv;
      }
    }
  }
}

// ---------------------------------------------------------------------------

extern "C" void kernel_launch(void* const* d_in, const int* in_sizes, int n_in,
                              void* d_out, int out_size, void* d_ws, size_t ws_size,
                              hipStream_t stream) {
  const float* seq  = (const float*)d_in[0];
  const float* attn = (const float*)d_in[1];
  const int*   pos  = (const int*)d_in[2];
  const int*   hts  = (const int*)d_in[3];
  const float* Wq   = (const float*)d_in[4];
  const float* bq   = (const float*)d_in[5];
  const float* Wk   = (const float*)d_in[6];
  const float* bk   = (const float*)d_in[7];
  const float* aw   = (const float*)d_in[8];
  const float* ab   = (const float*)d_in[9];
  const float* Wh   = (const float*)d_in[10];
  const float* bh   = (const float*)d_in[11];
  const float* Wt   = (const float*)d_in[12];
  const float* bt   = (const float*)d_in[13];
  const float* Wb   = (const float*)d_in[14];
  const float* bb   = (const float*)d_in[15];
  float* out = (float*)d_out;

  char* ws = (char*)d_ws;
  size_t off = 0;
  auto carve = [&](size_t bytes) -> char* {
    char* p = ws + off;
    off = (off + bytes + 255) & ~(size_t)255;
    return p;
  };
  u16*   seq_bf  = (u16*)  carve((size_t)NN*CC*DD*2);
  u16*   seqT_bf = (u16*)  carve((size_t)NN*CC*DD*2);
  u16*   WkT     = (u16*)  carve((size_t)DD*DD*2);
  u16*   WhT     = (u16*)  carve((size_t)DD*K3*2);
  u16*   WtT     = (u16*)  carve((size_t)DD*K3*2);
  u16*   WbT     = (u16*)  carve((size_t)NLL*KBIG*2);
  float* w1buf   = (float*)carve((DD + 1) * 4);
  float* ent_emb = (float*)carve((size_t)NN*EE*DD*4);
  float* ent_att = (float*)carve((size_t)NN*EE*HH*CC*4);
  u16*   ht_bf   = (u16*)  carve((size_t)NP*CC*2);
  float* rs      = (float*)carve((size_t)NP*DD*4);
  u16*   rs_bf   = (u16*)  carve((size_t)NP*DD*2);
  int*   arow_h  = (int*)  carve((size_t)NN*PPP*MMM*4);
  int*   arow_t  = (int*)  carve((size_t)NN*PPP*MMM*4);
  float* sk      = (float*)carve((size_t)NP*MMM*DD*4);
  float* hess    = (float*)carve((size_t)NP*DD*4);
  float* tess    = (float*)carve((size_t)NP*DD*4);
  u16*   cat_bf  = (u16*)  carve((size_t)NP*K3*2);
  u16*   zh_bf   = (u16*)  carve((size_t)NP*DD*2);
  u16*   zt_bf   = (u16*)  carve((size_t)NP*DD*2);

  auto cdiv = [](long a, long b) { return (int)((a + b - 1) / b); };

  // ---- precision staging ----
  k_cvt_bf16<<<cdiv((long)NN*CC*DD, 256), 256, 0, stream>>>(seq, seq_bf, (long)NN*CC*DD);
  k_seqT   <<<cdiv((long)NN*CC*DD, 256), 256, 0, stream>>>(seq, seqT_bf);
  k_tconv  <<<cdiv((long)DD*DD,   256), 256, 0, stream>>>(Wk, WkT, DD, DD);
  k_tconv  <<<cdiv((long)K3*DD,   256), 256, 0, stream>>>(Wh, WhT, K3, DD);
  k_tconv  <<<cdiv((long)K3*DD,   256), 256, 0, stream>>>(Wt, WtT, K3, DD);
  k_tconv  <<<cdiv((long)KBIG*NLL,256), 256, 0, stream>>>(Wb, WbT, KBIG, NLL);
  k_w1     <<<4, 256, 0, stream>>>(Wq, bq, aw, ab, w1buf);

  // ---- gathers / reductions (bandwidth stages) ----
  k_ent_emb<<<cdiv((long)NN*EE*DD, 256), 256, 0, stream>>>(seq, pos, ent_emb);
  k_ent_att<<<cdiv((long)NN*EE*HH*CC, 256), 256, 0, stream>>>(attn, pos, ent_att);
  k_ht     <<<NP, 256, 0, stream>>>(ent_att, hts, ht_bf);
  k_arow   <<<cdiv(2L*NN*PPP*MMM, 256), 256, 0, stream>>>(pos, hts, arow_h, arow_t);

  // ---- rs = ht_att @ seq  (batched over n via grid.y) ----
  {
    int tilesNg = DD / 64, tilesM = PPP / 16;
    dim3 grid(cdiv((long)tilesM * tilesNg, 8), NN);
    k_wmma_gemm<<<grid, 256, 0, stream>>>(ht_bf, nullptr, CC, (long)PPP*CC,
                                          seqT_bf, CC, (long)DD*CC,
                                          nullptr, rs, rs_bf, (long)PPP*DD,
                                          PPP, DD, CC, 0, tilesNg);
  }

  // ---- mention_guide head side: sk = h_embs @ Wk + bk, then softmax-mix ----
  {
    int tilesNg = DD / 64, tilesM = (NN*PPP*MMM) / 16;
    k_wmma_gemm<<<cdiv((long)tilesM * tilesNg, 8), 256, 0, stream>>>(
        seq_bf, arow_h, DD, 0, WkT, DD, 0, bk, sk, nullptr, 0,
        NN*PPP*MMM, DD, DD, 0, tilesNg);
    k_score<<<NP, 256, 0, stream>>>(rs, sk, w1buf, aw, pos, hts, 0, hess);
  }
  // ---- tail side (reuses sk buffer) ----
  {
    int tilesNg = DD / 64, tilesM = (NN*PPP*MMM) / 16;
    k_wmma_gemm<<<cdiv((long)tilesM * tilesNg, 8), 256, 0, stream>>>(
        seq_bf, arow_t, DD, 0, WkT, DD, 0, bk, sk, nullptr, 0,
        NN*PPP*MMM, DD, DD, 0, tilesNg);
    k_score<<<NP, 256, 0, stream>>>(rs, sk, w1buf, aw, pos, hts, 1, tess);
  }

  // ---- zh = tanh(tanh(cat_h @ Wh + bh)); zt = tanh(cat_t @ Wt + bt) ----
  {
    int tilesNg = DD / 64, tilesM = NP / 16;
    k_cat<<<cdiv((long)NP*K3, 256), 256, 0, stream>>>(rs, ent_emb, hess, hts, 0, cat_bf);
    k_wmma_gemm<<<cdiv((long)tilesM * tilesNg, 8), 256, 0, stream>>>(
        cat_bf, nullptr, K3, 0, WhT, K3, 0, bh, nullptr, zh_bf, 0,
        NP, DD, K3, 2, tilesNg);
    k_cat<<<cdiv((long)NP*K3, 256), 256, 0, stream>>>(rs, ent_emb, tess, hts, 1, cat_bf);
    k_wmma_gemm<<<cdiv((long)tilesM * tilesNg, 8), 256, 0, stream>>>(
        cat_bf, nullptr, K3, 0, WtT, K3, 0, bt, nullptr, zt_bf, 0,
        NP, DD, K3, 1, tilesNg);
  }

  // ---- fused bilinear classifier: out = bl @ Wb + bb ----
  {
    int tilesM = NP / 16;                         // 64 ; 2 column groups
    k_bilinear_out<<<cdiv((long)tilesM * 2, 8), 256, 0, stream>>>(
        zh_bf, zt_bf, WbT, bb, out);
  }
}
